// lstmVAE_initfeed_68624987456257
// MI455X (gfx1250) — matrix operand order repaired and checked by reading, not compiled, not run on hardware
//
#include <hip/hip_runtime.h>
#include <hip/hip_bf16.h>
#include <stdint.h>

#define B_   256
#define T_   256
#define DIM_ 96
#define DZ_  256
#define H_   512
#define G4H_ (4 * H_)

typedef __bf16 bf16_t;
typedef __bf16 v16bf __attribute__((ext_vector_type(16)));
typedef float  v8f   __attribute__((ext_vector_type(8)));

// ---------------------------------------------------------------- helpers ---

__device__ __forceinline__ v8f wmma_bf16(v16bf a, v16bf b, v8f c) {
  // D(16x16,f32) += A(16x32,bf16) * B(32x16,bf16)
  return __builtin_amdgcn_wmma_f32_16x16x32_bf16(false, a, false, b,
                                                 (short)0, c, false, false);
}

__device__ __forceinline__ v8f splat8(float x) {
  v8f r;
#pragma unroll
  for (int i = 0; i < 8; i++) r[i] = x;
  return r;
}

// A fragment: 16 rows x 32 K. lanes 0-15 rows, lane-half selects K sub-blocks
// {k0+half*8 .. +7} and {k0+16+half*8 .. +7} (per ISA 16-bit A layout).
__device__ __forceinline__ v16bf load_frag_a(const bf16_t* base, size_t ld,
                                             int row0, int k0, int lane) {
  const bf16_t* p = base + (size_t)(row0 + (lane & 15)) * ld +
                    (size_t)k0 + (size_t)((lane >> 4) << 3);
  v16bf r;
  ((uint4*)&r)[0] = *(const uint4*)p;
  ((uint4*)&r)[1] = *(const uint4*)(p + 16);
  return r;
}

// B fragment from W stored (N,K) row-major: lane n = column, lane-half selects K
// sub-blocks {k0+half*8 .. +7} and {k0+16+half*8 .. +7} (matches A layout).
__device__ __forceinline__ v16bf load_frag_b(const bf16_t* base, size_t ld,
                                             int n0, int k0, int lane) {
  const bf16_t* p = base + (size_t)(n0 + (lane & 15)) * ld +
                    (size_t)k0 + (size_t)((lane >> 4) << 3);
  v16bf r;
  ((uint4*)&r)[0] = *(const uint4*)p;
  ((uint4*)&r)[1] = *(const uint4*)(p + 16);
  return r;
}

__device__ __forceinline__ float sigm(float x) { return 1.f / (1.f + __expf(-x)); }

// Device-scope grid barrier over nWG workgroups using a monotonic counter.
__device__ __forceinline__ void grid_bar(unsigned* cnt, unsigned target) {
  __threadfence();
  __syncthreads();
  if (threadIdx.x == 0) {
    __hip_atomic_fetch_add(cnt, 1u, __ATOMIC_ACQ_REL, __HIP_MEMORY_SCOPE_AGENT);
    while (__hip_atomic_load(cnt, __ATOMIC_ACQUIRE, __HIP_MEMORY_SCOPE_AGENT) < target)
      __builtin_amdgcn_s_sleep(1);
  }
  __syncthreads();
}

// ---------------------------------------------------------------- utility ---

__global__ void k_zero32(unsigned* p, int n) {
  int i = blockIdx.x * 256 + threadIdx.x;
  if (i < n) p[i] = 0u;
}

__global__ void k_cvt(const float* __restrict__ in, bf16_t* __restrict__ out, int n) {
  int i = blockIdx.x * 256 + threadIdx.x;
  if (i < n) out[i] = (bf16_t)in[i];
}

// per-batch layernorm over (T,DIM) with per-(t,d) gamma/beta -> bf16
__global__ void k_ln(const float* __restrict__ x, const float* __restrict__ g,
                     const float* __restrict__ bta, bf16_t* __restrict__ xn) {
  __shared__ float s0[256], s1[256];
  const int b = blockIdx.x, tid = threadIdx.x;
  const float* xb = x + (size_t)b * T_ * DIM_;
  float s = 0.f, q = 0.f;
  for (int i = tid; i < T_ * DIM_; i += 256) { float v = xb[i]; s += v; q += v * v; }
  s0[tid] = s; s1[tid] = q;
  __syncthreads();
  for (int st = 128; st > 0; st >>= 1) {
    if (tid < st) { s0[tid] += s0[tid + st]; s1[tid] += s1[tid + st]; }
    __syncthreads();
  }
  const float inv_n = 1.f / (float)(T_ * DIM_);
  const float mean = s0[0] * inv_n;
  const float var  = s1[0] * inv_n - mean * mean;
  const float rstd = rsqrtf(var + 1e-5f);
  for (int i = tid; i < T_ * DIM_; i += 256)
    xn[(size_t)b * T_ * DIM_ + i] = (bf16_t)((xb[i] - mean) * rstd * g[i] + bta[i]);
}

// ------------------------------------------------------- generic WMMA GEMM ---
// C[M,N] = act(A[M,K] @ W[N,K]^T + bias). Block tile 128x64, wave tile 32x32.
// Requires M%128==0, N%64==0, K%32==0.
__global__ void __launch_bounds__(256) k_gemm(
    const bf16_t* __restrict__ A, int lda,
    const bf16_t* __restrict__ W, int ldw,
    const float* __restrict__ bias,
    bf16_t* __restrict__ outb, float* __restrict__ outf,
    int ldo, int K, int relu) {
  const int lane = threadIdx.x & 31, wid = threadIdx.x >> 5;
  const int gm0 = blockIdx.y * 128 + (wid & 3) * 32;
  const int gn0 = blockIdx.x * 64  + (wid >> 2) * 32;
  v8f c00 = splat8(0.f), c01 = splat8(0.f), c10 = splat8(0.f), c11 = splat8(0.f);
  for (int kk = 0; kk < K; kk += 32) {
    v16bf a0 = load_frag_a(A, (size_t)lda, gm0,      kk, lane);
    v16bf a1 = load_frag_a(A, (size_t)lda, gm0 + 16, kk, lane);
    v16bf b0 = load_frag_b(W, (size_t)ldw, gn0,      kk, lane);
    v16bf b1 = load_frag_b(W, (size_t)ldw, gn0 + 16, kk, lane);
    c00 = wmma_bf16(a0, b0, c00);
    c01 = wmma_bf16(a0, b1, c01);
    c10 = wmma_bf16(a1, b0, c10);
    c11 = wmma_bf16(a1, b1, c11);
    __builtin_prefetch(A + (size_t)(gm0 + (lane & 15)) * lda + kk + 64, 0, 1);
  }
  const int col = lane & 15, lh = lane >> 4;
  auto st = [&](v8f c, int mb, int nb) {
    float bi = bias ? bias[nb + col] : 0.f;
#pragma unroll
    for (int v = 0; v < 8; v++) {
      float val = c[v] + bi;
      if (relu) val = fmaxf(val, 0.f);
      size_t o = (size_t)(mb + lh * 8 + v) * (size_t)ldo + (size_t)(nb + col);
      if (outb) outb[o] = (bf16_t)val;
      if (outf) outf[o] = val;
    }
  };
  st(c00, gm0, gn0); st(c01, gm0, gn0 + 16);
  st(c10, gm0 + 16, gn0); st(c11, gm0 + 16, gn0 + 16);
}

// --------------------------------------------------- persistent biLSTM ------
// grid = 64 WGs x 256 thr. WGs 0-31: forward, 32-63: backward.
// WG w owns h columns [16w,16w+16); wave j owns batch rows [32j,32j+32).
// Per step: gates = h_prev @ Whh^T + x_t @ Wih^T + b  (K = 512 + 256),
// weight slice (64 rows x 768 K, 96KB) staged in LDS once.
__global__ void __launch_bounds__(256) k_bilstm(
    const bf16_t* __restrict__ hin,   // (B,T,DZ) bf16 (encoder output)
    const bf16_t* __restrict__ Wihf, const bf16_t* __restrict__ Whhf,
    const bf16_t* __restrict__ Wihb, const bf16_t* __restrict__ Whhb,
    const float* __restrict__ biasf, const float* __restrict__ biasb,
    const int* __restrict__ lens,
    bf16_t* __restrict__ hbuf,        // [dir][pp][B][H] bf16 (pp0 pre-zeroed)
    bf16_t* __restrict__ pooled,      // (B, 2H) bf16
    unsigned* __restrict__ bars) {
  extern __shared__ bf16_t sW[];      // 64 rows x 768 K
  const int dir  = blockIdx.x >> 5;
  const int w    = blockIdx.x & 31;
  const int tid  = threadIdx.x, lane = tid & 31, wave = tid >> 5;
  const int hc0  = w * 16, m0 = wave * 32;
  const int col  = lane & 15, lh = lane >> 4;
  const bf16_t* Wih = dir ? Wihb : Wihf;
  const bf16_t* Whh = dir ? Whhb : Whhf;
  const float*  bias = dir ? biasb : biasf;
  bf16_t* hb = hbuf + (size_t)dir * 2 * B_ * H_;
  unsigned* bar = bars + dir;

  // Stage gate-weight slice: row r = g*16+nl -> W row (g*H + hc0 + nl);
  // K 0..511 from Whh, K 512..767 from Wih.
  for (int idx = tid; idx < 64 * 96; idx += 256) {
    int r = idx / 96, k = (idx % 96) * 8;
    int g = r >> 4, nl = r & 15;
    int wrow = g * H_ + hc0 + nl;
    uint4 v;
    if (k < H_) v = *(const uint4*)(Whh + (size_t)wrow * H_  + k);
    else        v = *(const uint4*)(Wih + (size_t)wrow * DZ_ + (k - H_));
    *(uint4*)(sW + (size_t)r * 768 + k) = v;
  }
  __syncthreads();

  float bgate[4];
#pragma unroll
  for (int g = 0; g < 4; g++) bgate[g] = bias[g * H_ + hc0 + col];
  int lr[16];
#pragma unroll
  for (int mi = 0; mi < 2; mi++)
#pragma unroll
    for (int v = 0; v < 8; v++) lr[mi * 8 + v] = lens[m0 + mi * 16 + lh * 8 + v];

  float cst[16], hreg[16], pool[16];
#pragma unroll
  for (int i = 0; i < 16; i++) { cst[i] = 0.f; hreg[i] = 0.f; pool[i] = 0.f; }

  unsigned target = 0;
  for (int t = 0; t < T_; ++t) {
    const int tt = dir ? (T_ - 1 - t) : t;
    const bf16_t* hcur = hb + (size_t)(t & 1) * (B_ * H_);
    bf16_t*       hnxt = hb + (size_t)((t + 1) & 1) * (B_ * H_);

    v8f acc[2][4];
#pragma unroll
    for (int mi = 0; mi < 2; mi++)
#pragma unroll
      for (int g = 0; g < 4; g++) acc[mi][g] = splat8(bgate[g]);

    // recurrent part: K = 512, A = h_prev (bf16, B x H)
    for (int kk = 0; kk < H_; kk += 32) {
      v16bf a0 = load_frag_a(hcur, H_, m0,      kk, lane);
      v16bf a1 = load_frag_a(hcur, H_, m0 + 16, kk, lane);
#pragma unroll
      for (int g = 0; g < 4; g++) {
        v16bf bfr = load_frag_b(sW, 768, g * 16, kk, lane);
        acc[0][g] = wmma_bf16(a0, bfr, acc[0][g]);
        acc[1][g] = wmma_bf16(a1, bfr, acc[1][g]);
      }
    }
    // input part: K = 256, A rows are h_in[b][tt][:] (row stride T*DZ)
    const bf16_t* xbase = hin + (size_t)tt * DZ_;
    for (int kk = 0; kk < DZ_; kk += 32) {
      v16bf a0 = load_frag_a(xbase, (size_t)T_ * DZ_, m0,      kk, lane);
      v16bf a1 = load_frag_a(xbase, (size_t)T_ * DZ_, m0 + 16, kk, lane);
#pragma unroll
      for (int g = 0; g < 4; g++) {
        v16bf bfr = load_frag_b(sW, 768, g * 16, H_ + kk, lane);
        acc[0][g] = wmma_bf16(a0, bfr, acc[0][g]);
        acc[1][g] = wmma_bf16(a1, bfr, acc[1][g]);
      }
    }
    // nonlinearity + masked state update + pooling
#pragma unroll
    for (int mi = 0; mi < 2; mi++)
#pragma unroll
      for (int v = 0; v < 8; v++) {
        int idx = mi * 8 + v;
        float cn = sigm(acc[mi][1][v]) * cst[idx] +
                   sigm(acc[mi][0][v]) * tanhf(acc[mi][2][v]);
        float hn = sigm(acc[mi][3][v]) * tanhf(cn);
        bool m = tt < lr[idx];
        float hnew = m ? hn : hreg[idx];
        cst[idx]  = m ? cn : cst[idx];
        pool[idx] += m ? hn : 0.f;
        hreg[idx] = hnew;
        int brow = m0 + mi * 16 + lh * 8 + v;
        hnxt[(size_t)brow * H_ + hc0 + col] = (bf16_t)hnew;
      }
    target += 32;
    grid_bar(bar, target);
  }
  // pooled = sum(masked h) / len
#pragma unroll
  for (int mi = 0; mi < 2; mi++)
#pragma unroll
    for (int v = 0; v < 8; v++) {
      int idx = mi * 8 + v;
      int brow = m0 + mi * 16 + lh * 8 + v;
      pooled[(size_t)brow * (2 * H_) + (size_t)dir * H_ + hc0 + col] =
          (bf16_t)(pool[idx] / (float)lr[idx]);
    }
}

// ------------------------------------------------- misc small elementwise ---

__global__ void k_z(const float* __restrict__ eps, const float* __restrict__ mu,
                    const float* __restrict__ lv, float* __restrict__ zf,
                    bf16_t* __restrict__ zb) {
  int i = blockIdx.x * 256 + threadIdx.x;
  float z = eps[i] * __expf(0.5f * lv[i]) + mu[i];
  zf[i] = z;
  zb[i] = (bf16_t)z;
}

__global__ void k_lenest(const bf16_t* __restrict__ e1t, const float* __restrict__ w,
                         const float* __restrict__ b, float* __restrict__ out) {
  int i = threadIdx.x;  // one block of 256
  float acc = b[0];
#pragma unroll
  for (int j = 0; j < 64; j++) acc += (float)e1t[i * 64 + j] * w[j];
  out[i] = acc;
}

__global__ void k_decinit(const float* __restrict__ x, bf16_t* __restrict__ pose) {
  int i = blockIdx.x * 256 + threadIdx.x;
  if (i < B_ * DIM_) {
    int b = i / DIM_, d = i % DIM_;
    pose[i] = (bf16_t)x[(size_t)b * T_ * DIM_ + d];
  }
}

// --------------------------------------------------- persistent decoder -----
// grid = 32 WGs x 256 thr. Per step 3 phases with grid barriers:
//  1) gates = Gxz + pose @ dWihP^T + h @ dWhh^T -> hn   (all 32 WGs)
//  2) h_ = relu(hn @ d1W^T + d1b)                       (WGs 0-15)
//  3) pose' = h_ @ d2W^T + d2b ; recon[:,t,:] = pose'   (WGs 0-5)
__global__ void __launch_bounds__(256) k_decoder(
    const bf16_t* __restrict__ Gxz,   // (B, 4H) = xz @ dWih[:, :DZ]^T + db
    const bf16_t* __restrict__ dWih,  // (4H, 352) bf16
    const bf16_t* __restrict__ dWhh,  // (4H, 512) bf16
    const bf16_t* __restrict__ d1W, const float* __restrict__ d1b,
    const bf16_t* __restrict__ d2W, const float* __restrict__ d2b,
    bf16_t* __restrict__ hbuf,        // 2 x (B,H) ping-pong (pp0 pre-zeroed)
    bf16_t* __restrict__ hmid,        // (B,256)
    bf16_t* __restrict__ pose,        // (B,96)
    float* __restrict__ recon,        // (B,T,96) f32 (d_out slice)
    unsigned* __restrict__ bar) {
  extern __shared__ bf16_t sW[];      // 64 rows x 608 K
  const int w = blockIdx.x, tid = threadIdx.x;
  const int lane = tid & 31, wave = tid >> 5;
  const int hc0 = w * 16, m0 = wave * 32;
  const int col = lane & 15, lh = lane >> 4;

  // stage phase-1 weights: K 0..511 from dWhh, K 512..607 = pose cols of dWih
  for (int idx = tid; idx < 64 * 76; idx += 256) {
    int r = idx / 76, k = (idx % 76) * 8;
    int g = r >> 4, nl = r & 15;
    int wrow = g * H_ + hc0 + nl;
    uint4 v;
    if (k < H_) v = *(const uint4*)(dWhh + (size_t)wrow * H_ + k);
    else        v = *(const uint4*)(dWih + (size_t)wrow * 352 + DZ_ + (k - H_));
    *(uint4*)(sW + (size_t)r * 608 + k) = v;
  }
  __syncthreads();

  float ginit[64];
#pragma unroll
  for (int mi = 0; mi < 2; mi++)
#pragma unroll
    for (int g = 0; g < 4; g++)
#pragma unroll
      for (int v = 0; v < 8; v++)
        ginit[(mi * 4 + g) * 8 + v] =
            (float)Gxz[(size_t)(m0 + mi * 16 + lh * 8 + v) * G4H_ + g * H_ + hc0 + col];

  float cst[16];
#pragma unroll
  for (int i = 0; i < 16; i++) cst[i] = 0.f;

  unsigned target = 0;
  for (int t = 0; t < T_; ++t) {
    bf16_t* hcur = hbuf + (size_t)(t & 1) * (B_ * H_);
    bf16_t* hnxt = hbuf + (size_t)((t + 1) & 1) * (B_ * H_);

    // ---- phase 1: gates + state
    v8f acc[2][4];
#pragma unroll
    for (int mi = 0; mi < 2; mi++)
#pragma unroll
      for (int g = 0; g < 4; g++) acc[mi][g] = splat8(ginit[(mi * 4 + g) * 8]);
    // (ginit varies with v; fix up by adding residual below)
#pragma unroll
    for (int mi = 0; mi < 2; mi++)
#pragma unroll
      for (int g = 0; g < 4; g++)
#pragma unroll
        for (int v = 0; v < 8; v++) acc[mi][g][v] = ginit[(mi * 4 + g) * 8 + v];

    for (int kk = 0; kk < H_; kk += 32) {
      v16bf a0 = load_frag_a(hcur, H_, m0,      kk, lane);
      v16bf a1 = load_frag_a(hcur, H_, m0 + 16, kk, lane);
#pragma unroll
      for (int g = 0; g < 4; g++) {
        v16bf bfr = load_frag_b(sW, 608, g * 16, kk, lane);
        acc[0][g] = wmma_bf16(a0, bfr, acc[0][g]);
        acc[1][g] = wmma_bf16(a1, bfr, acc[1][g]);
      }
    }
    for (int kk = 0; kk < DIM_; kk += 32) {
      v16bf a0 = load_frag_a(pose, DIM_, m0,      kk, lane);
      v16bf a1 = load_frag_a(pose, DIM_, m0 + 16, kk, lane);
#pragma unroll
      for (int g = 0; g < 4; g++) {
        v16bf bfr = load_frag_b(sW, 608, g * 16, H_ + kk, lane);
        acc[0][g] = wmma_bf16(a0, bfr, acc[0][g]);
        acc[1][g] = wmma_bf16(a1, bfr, acc[1][g]);
      }
    }
#pragma unroll
    for (int mi = 0; mi < 2; mi++)
#pragma unroll
      for (int v = 0; v < 8; v++) {
        int idx = mi * 8 + v;
        float cn = sigm(acc[mi][1][v]) * cst[idx] +
                   sigm(acc[mi][0][v]) * tanhf(acc[mi][2][v]);
        float hn = sigm(acc[mi][3][v]) * tanhf(cn);
        cst[idx] = cn;
        hnxt[(size_t)(m0 + mi * 16 + lh * 8 + v) * H_ + hc0 + col] = (bf16_t)hn;
      }
    target += 32;
    grid_bar(bar, target);

    // ---- phase 2: h_ = relu(hn @ d1W^T + d1b), WGs 0..15 own 16 cols each
    if (w < 16) {
      int n0 = w * 16;
      v8f a2[2];
      a2[0] = splat8(0.f); a2[1] = splat8(0.f);
      for (int kk = 0; kk < H_; kk += 32) {
        v16bf a0 = load_frag_a(hnxt, H_, m0,      kk, lane);
        v16bf a1 = load_frag_a(hnxt, H_, m0 + 16, kk, lane);
        v16bf b0 = load_frag_b(d1W, H_, n0, kk, lane);
        a2[0] = wmma_bf16(a0, b0, a2[0]);
        a2[1] = wmma_bf16(a1, b0, a2[1]);
      }
      float bi = d1b[n0 + col];
#pragma unroll
      for (int mi = 0; mi < 2; mi++)
#pragma unroll
        for (int v = 0; v < 8; v++) {
          float val = fmaxf(a2[mi][v] + bi, 0.f);
          hmid[(size_t)(m0 + mi * 16 + lh * 8 + v) * DZ_ + n0 + col] = (bf16_t)val;
        }
    }
    target += 32;
    grid_bar(bar, target);

    // ---- phase 3: pose' = h_ @ d2W^T + d2b, WGs 0..5 own 16 cols each
    if (w < 6) {
      int n0 = w * 16;
      v8f a3[2];
      a3[0] = splat8(0.f); a3[1] = splat8(0.f);
      for (int kk = 0; kk < DZ_; kk += 32) {
        v16bf a0 = load_frag_a(hmid, DZ_, m0,      kk, lane);
        v16bf a1 = load_frag_a(hmid, DZ_, m0 + 16, kk, lane);
        v16bf b0 = load_frag_b(d2W, DZ_, n0, kk, lane);
        a3[0] = wmma_bf16(a0, b0, a3[0]);
        a3[1] = wmma_bf16(a1, b0, a3[1]);
      }
      float bi = d2b[n0 + col];
#pragma unroll
      for (int mi = 0; mi < 2; mi++)
#pragma unroll
        for (int v = 0; v < 8; v++) {
          int brow = m0 + mi * 16 + lh * 8 + v;
          float val = a3[mi][v] + bi;
          pose[(size_t)brow * DIM_ + n0 + col] = (bf16_t)val;
          recon[(size_t)brow * T_ * DIM_ + (size_t)t * DIM_ + n0 + col] = val;
        }
    }
    target += 32;
    grid_bar(bar, target);
  }
}

// ------------------------------------------------------------------ host ----

static const size_t X_ELEMS   = (size_t)B_ * T_ * DIM_;     // 6291456
static const size_t RECON_OFF = X_ELEMS;
static const size_t LEN_OFF   = RECON_OFF + X_ELEMS;        // 12582912
static const size_t MU_OFF    = LEN_OFF + B_;
static const size_t LV_OFF    = MU_OFF + (size_t)B_ * DZ_;
static const size_t Z_OFF     = LV_OFF + (size_t)B_ * DZ_;
static const size_t LE_OFF    = Z_OFF + (size_t)B_ * DZ_;

extern "C" void kernel_launch(void* const* d_in, const int* in_sizes, int n_in,
                              void* d_out, int out_size, void* d_ws, size_t ws_size,
                              hipStream_t stream) {
  (void)in_sizes; (void)n_in; (void)out_size; (void)ws_size;
  char* ws = (char*)d_ws;
  size_t off = 0;
  auto alloc = [&](size_t bytes) -> char* {
    char* p = ws + off;
    off = (off + bytes + 255) & ~(size_t)255;
    return p;
  };

  const float* x      = (const float*)d_in[0];
  const int*   xlen   = (const int*)d_in[1];
  const float* eps    = (const float*)d_in[2];
  float*       outF   = (float*)d_out;

  // ---- bf16 weight conversions
  struct WInfo { int idx; size_t elems; };
  static const WInfo wl[15] = {
      {5, 256 * 96},     {7, 2048 * 256},  {8, 2048 * 512},
      {10, 2048 * 256},  {11, 2048 * 512}, {13, 512 * 1024},
      {15, 256 * 512},   {17, 512 * 1024}, {19, 256 * 512},
      {21, 256 * 256},   {23, 2048 * 352}, {24, 2048 * 512},
      {26, 256 * 512},   {28, 96 * 256},   {30, 64 * 256}};
  bf16_t* wb[15];
  for (int i = 0; i < 15; i++) wb[i] = (bf16_t*)alloc(wl[i].elems * sizeof(bf16_t));
  bf16_t* encWb = wb[0];  bf16_t* Wihf = wb[1];  bf16_t* Whhf = wb[2];
  bf16_t* Wihb = wb[3];   bf16_t* Whhb = wb[4];  bf16_t* mu1b = wb[5];
  bf16_t* mu2b = wb[6];   bf16_t* lv1b = wb[7];  bf16_t* lv2b = wb[8];
  bf16_t* decb = wb[9];   bf16_t* dWihB = wb[10]; bf16_t* dWhhB = wb[11];
  bf16_t* d1Wb = wb[12];  bf16_t* d2Wb = wb[13]; bf16_t* e1Wb = wb[14];

  // ---- activations / state
  bf16_t* xn     = (bf16_t*)alloc(X_ELEMS * sizeof(bf16_t));            // (B*T, 96)
  bf16_t* hin    = (bf16_t*)alloc((size_t)B_ * T_ * DZ_ * sizeof(bf16_t)); // (B,T,DZ)
  bf16_t* hbufE  = (bf16_t*)alloc((size_t)4 * B_ * H_ * sizeof(bf16_t)); // [dir][pp]
  bf16_t* pooled = (bf16_t*)alloc((size_t)B_ * 2 * H_ * sizeof(bf16_t));
  bf16_t* t1     = (bf16_t*)alloc((size_t)B_ * H_ * sizeof(bf16_t));
  bf16_t* t2     = (bf16_t*)alloc((size_t)B_ * H_ * sizeof(bf16_t));
  bf16_t* zbf    = (bf16_t*)alloc((size_t)B_ * DZ_ * sizeof(bf16_t));
  bf16_t* xzbf   = (bf16_t*)alloc((size_t)B_ * DZ_ * sizeof(bf16_t));
  bf16_t* e1t    = (bf16_t*)alloc((size_t)B_ * 64 * sizeof(bf16_t));
  bf16_t* Gxz    = (bf16_t*)alloc((size_t)B_ * G4H_ * sizeof(bf16_t));
  bf16_t* hbufD  = (bf16_t*)alloc((size_t)2 * B_ * H_ * sizeof(bf16_t));
  bf16_t* hmid   = (bf16_t*)alloc((size_t)B_ * DZ_ * sizeof(bf16_t));
  bf16_t* pose   = (bf16_t*)alloc((size_t)B_ * DIM_ * sizeof(bf16_t));
  unsigned* bars = (unsigned*)alloc(256);

  // ---- init: barriers + h ping-pong buffers
  k_zero32<<<1, 256, 0, stream>>>(bars, 8);
  k_zero32<<<(int)((4 * B_ * H_ * 2 / 4 + 255) / 256), 256, 0, stream>>>(
      (unsigned*)hbufE, 4 * B_ * H_ * 2 / 4);
  k_zero32<<<(int)((2 * B_ * H_ * 2 / 4 + 255) / 256), 256, 0, stream>>>(
      (unsigned*)hbufD, 2 * B_ * H_ * 2 / 4);

  // ---- weight conversion
  for (int i = 0; i < 15; i++) {
    int n = (int)wl[i].elems;
    k_cvt<<<(n + 255) / 256, 256, 0, stream>>>((const float*)d_in[wl[i].idx], wb[i], n);
  }

  // ---- layernorm
  k_ln<<<B_, 256, 0, stream>>>(x, (const float*)d_in[3], (const float*)d_in[4], xn);

  auto gemm = [&](const bf16_t* A, int lda, const bf16_t* W, int ldw,
                  const float* bias, bf16_t* ob, float* of, int ldo,
                  int M, int N, int K, int relu) {
    dim3 g(N / 64, M / 128);
    k_gemm<<<g, 256, 0, stream>>>(A, lda, W, ldw, bias, ob, of, ldo, K, relu);
  };

  // ---- encoder linear: h_in = relu(xn @ encW^T + b)   [65536 x 256, K=96]
  gemm(xn, DIM_, encWb, DIM_, (const float*)d_in[6], hin, nullptr, DZ_,
       B_ * T_, DZ_, DIM_, 1);

  // ---- persistent bidirectional LSTM (64 WGs, LDS-staged weights)
  {
    size_t shmem = (size_t)64 * 768 * sizeof(bf16_t);  // 96 KB
    hipFuncSetAttribute(reinterpret_cast<const void*>(k_bilstm),
                        hipFuncAttributeMaxDynamicSharedMemorySize, (int)shmem);
    k_bilstm<<<64, 256, shmem, stream>>>(hin, Wihf, Whhf, Wihb, Whhb,
                                         (const float*)d_in[9], (const float*)d_in[12],
                                         xlen, hbufE, pooled, bars);
  }

  // ---- mu / log_var heads
  gemm(pooled, 2 * H_, mu1b, 2 * H_, (const float*)d_in[14], t1, nullptr, H_,
       B_, H_, 2 * H_, 1);
  gemm(t1, H_, mu2b, H_, (const float*)d_in[16], nullptr, outF + MU_OFF, DZ_,
       B_, DZ_, H_, 0);
  gemm(pooled, 2 * H_, lv1b, 2 * H_, (const float*)d_in[18], t2, nullptr, H_,
       B_, H_, 2 * H_, 1);
  gemm(t2, H_, lv2b, H_, (const float*)d_in[20], nullptr, outF + LV_OFF, DZ_,
       B_, DZ_, H_, 0);

  // ---- z = eps * exp(0.5*lv) + mu
  k_z<<<(B_ * DZ_) / 256, 256, 0, stream>>>(eps, outF + MU_OFF, outF + LV_OFF,
                                            outF + Z_OFF, zbf);

  // ---- xz = relu(z @ dec_lin^T + b) ; length head
  gemm(zbf, DZ_, decb, DZ_, (const float*)d_in[22], xzbf, nullptr, DZ_,
       B_, DZ_, DZ_, 1);
  gemm(zbf, DZ_, e1Wb, DZ_, (const float*)d_in[31], e1t, nullptr, 64,
       B_, 64, DZ_, 1);
  k_lenest<<<1, 256, 0, stream>>>(e1t, (const float*)d_in[32],
                                  (const float*)d_in[33], outF + LE_OFF);

  // ---- decoder: pose0 = x[:,0,:]; Gxz = xz @ dWih[:, :DZ]^T + db
  k_decinit<<<(B_ * DIM_ + 255) / 256, 256, 0, stream>>>(x, pose);
  gemm(xzbf, DZ_, dWihB, 352, (const float*)d_in[25], Gxz, nullptr, G4H_,
       B_, G4H_, DZ_, 0);

  {
    size_t shmem = (size_t)64 * 608 * sizeof(bf16_t);  // ~76 KB
    hipFuncSetAttribute(reinterpret_cast<const void*>(k_decoder),
                        hipFuncAttributeMaxDynamicSharedMemorySize, (int)shmem);
    k_decoder<<<32, 256, shmem, stream>>>(Gxz, dWihB, dWhhB, d1Wb,
                                          (const float*)d_in[27], d2Wb,
                                          (const float*)d_in[29], hbufD, hmid,
                                          pose, outF + RECON_OFF, bars + 2);
  }

  // ---- passthrough outputs: x and x_lengths (raw bits into float buffer)
  hipMemcpyAsync(outF, x, X_ELEMS * sizeof(float), hipMemcpyDeviceToDevice, stream);
  hipMemcpyAsync((char*)d_out + LEN_OFF * sizeof(float), xlen, B_ * sizeof(int),
                 hipMemcpyDeviceToDevice, stream);
}
